// UnconditionalModel_5772436045995
// MI455X (gfx1250) — matrix operand order repaired and checked by reading, not compiled, not run on hardware
//
#include <hip/hip_runtime.h>
#include <hip/hip_bf16.h>

typedef __attribute__((ext_vector_type(16))) _Float16 v16h;
typedef __attribute__((ext_vector_type(8)))  _Float16 v8h;
typedef __attribute__((ext_vector_type(8)))  float    v8f;
typedef __attribute__((ext_vector_type(4)))  int      v4i;

#define NWG       50
#define TPB       128              // 4 waves / WG
#define NTHREADS  (NWG * TPB)      // 6400 threads, 200 waves

#define Ssz   700
#define G4H   1600
#define NT    100                  // 1600 / 16 column tiles
#define KP1   416                  // layer1 padded K (3 + 400 -> 416)
#define KC1   13
#define KP2   832                  // layers 2/3 padded K (3 + 400 + 400 -> 832)
#define KC2   26
#define NHEAD 128                  // head N padded (121 -> 128)
#define KCH   13
#define KPH   416                  // head K padded (400 -> 416)

// -------- LDS layout (halves): per-WG weight slices, fragment-ordered --------
#define LDS_L1   0                       // 13*2*512 = 13312
#define LDS_L2   (LDS_L1 + KC1 * 1024)   // 26*2*512 = 26624
#define LDS_L3   (LDS_L2 + KC2 * 1024)
#define LDS_H    (LDS_L3 + KC2 * 1024)   // 13*512 = 6656 (WGs 0..7 only)
#define LDS_HALVES (LDS_H + KCH * 512)   // 73216 halves
#define LDS_BYTES  (LDS_HALVES * 2)      // 146432 B

// -------- output offsets (floats): e,pi,mu1,mu2,sig1,sig2,ro,h1,c1,h2,c2,h3,c3
#define OFF_E    0
#define OFF_PI   44800
#define OFF_MU1  (44800 + 896000 * 1)
#define OFF_MU2  (44800 + 896000 * 2)
#define OFF_SIG1 (44800 + 896000 * 3)
#define OFF_SIG2 (44800 + 896000 * 4)
#define OFF_RO   (44800 + 896000 * 5)
#define OFF_H1   (44800 + 896000 * 6)
#define OFF_C1   (OFF_H1 + 25600)
#define OFF_H2   (OFF_H1 + 25600 * 2)
#define OFF_C2   (OFF_H1 + 25600 * 3)
#define OFF_H3   (OFF_H1 + 25600 * 4)
#define OFF_C3   (OFF_H1 + 25600 * 5)

// gfx1250 async global->LDS path (probe-confirmed present on this toolchain;
// signature: (v4i AS1* src, v4i AS3* dst, int offset, int cpol))
#if defined(__has_builtin)
#if __has_builtin(__builtin_amdgcn_global_load_async_to_lds_b128)
#define USE_ASYNC_LDS 1
#endif
#endif
#ifndef USE_ASYNC_LDS
#define USE_ASYNC_LDS 0
#endif

__device__ __forceinline__ void wait_async_zero() {
#if defined(__has_builtin)
#if __has_builtin(__builtin_amdgcn_s_wait_asynccnt)
    __builtin_amdgcn_s_wait_asynccnt(0);
    return;
#endif
#endif
    asm volatile("s_wait_asynccnt 0" ::: "memory");
}

__device__ __forceinline__ float sigmoidf_fast(float x) {
    return 1.0f / (1.0f + __expf(-x));
}

// ---------------------------------------------------------------------------
// A 16x32 f16 fragment: lane<16 holds K {0..15},
// lane>=16 holds K {16..31}; row = lane&15.
__device__ __forceinline__ v16h load_a_frag(const _Float16* __restrict__ act, int Kpad,
                                            int mt, int kc, int lane) {
    int m  = mt * 16 + (lane & 15);
    int k0 = kc * 32 + ((lane >> 4) << 4);
    const _Float16* p = act + m * Kpad + k0;
    v8h lo = *(const v8h*)(p);
    v8h hi = *(const v8h*)(p + 8);
    return __builtin_shufflevector(lo, hi, 0, 1, 2, 3, 4, 5, 6, 7,
                                           8, 9, 10, 11, 12, 13, 14, 15);
}

// ---------------------------------------------------------------------------
// Global sense-reversing barrier across NWG workgroups (bar[0]=count, bar[1]=generation)
__device__ __forceinline__ void gbar(int* bar) {
    __syncthreads();
    if (threadIdx.x == 0) {
        __threadfence();
        volatile int* vgen = (volatile int*)(bar + 1);
        int g = *vgen;
        if (atomicAdd(bar, 1) == NWG - 1) {
            bar[0] = 0;
            __threadfence();
            atomicAdd(bar + 1, 1);
        } else {
            while (*vgen == g) { __builtin_amdgcn_s_sleep(1); }
        }
    }
    __syncthreads();
}

// ---------------------------------------------------------------------------
// gates(64x1600) = act(64xKpad) @ W + bias.  B fragments come from LDS (weights
// staged once); A fragments are software-pipelined from global.
__device__ __forceinline__ void gemm_layer(const _Float16* __restrict__ act, int Kpad,
                                           const _Float16* __restrict__ ldsW, int sl,
                                           const float* __restrict__ bias,
                                           float* __restrict__ gates,
                                           int KC, int wave, int lane) {
    int nt  = wave >> 1;
    int mt0 = (wave & 1) << 1;
    v8f acc0 = {};
    v8f acc1 = {};
    v16h a0 = load_a_frag(act, Kpad, mt0,     0, lane);
    v16h a1 = load_a_frag(act, Kpad, mt0 + 1, 0, lane);
    for (int kc = 0; kc < KC; ++kc) {
        v16h bf = *(const v16h*)(ldsW + ((kc * 2 + sl) * 32 + lane) * 16);
        v16h a0n = a0, a1n = a1;
        if (kc + 1 < KC) {                      // prefetch next-k A fragments
            a0n = load_a_frag(act, Kpad, mt0,     kc + 1, lane);
            a1n = load_a_frag(act, Kpad, mt0 + 1, kc + 1, lane);
        }
        acc0 = __builtin_amdgcn_wmma_f32_16x16x32_f16(false, a0, false, bf, (short)0, acc0, false, false);
        acc1 = __builtin_amdgcn_wmma_f32_16x16x32_f16(false, a1, false, bf, (short)0, acc1, false, false);
        a0 = a0n;
        a1 = a1n;
    }
    int n     = nt * 16 + (lane & 15);
    float bv  = bias[n];
    int mbase = mt0 * 16 + ((lane >> 4) << 3);
#pragma unroll
    for (int r = 0; r < 8; ++r) {
        gates[(mbase + r)      * G4H + n] = acc0[r] + bv;
        gates[(mbase + 16 + r) * G4H + n] = acc1[r] + bv;
    }
}

// Elementwise LSTM cell update; write h (f16) to two activation slots.
__device__ __forceinline__ void lstm_ew(const float* __restrict__ gates,
                                        float* __restrict__ c, float* __restrict__ hf,
                                        _Float16* __restrict__ dA, int sA, int cA,
                                        _Float16* __restrict__ dB, int sB, int cB,
                                        int gtid) {
    for (int e = gtid; e < 64 * 400; e += NTHREADS) {
        int b = e / 400, j = e - b * 400;
        const float* g = gates + b * G4H;
        float ig = sigmoidf_fast(g[j]);
        float fg = sigmoidf_fast(g[400 + j]);
        float gg = tanhf(g[800 + j]);
        float og = sigmoidf_fast(g[1200 + j]);
        float cn = fg * c[e] + ig * gg;
        float hn = og * tanhf(cn);
        c[e] = cn;
        hf[e] = hn;
        _Float16 hh = (_Float16)hn;
        dA[b * sA + cA + j] = hh;
        dB[b * sB + cB + j] = hh;
    }
}

// Stage one weight slice (nstrips strips, fragment order) from global pack into LDS.
__device__ __forceinline__ void stage_slice(const _Float16* __restrict__ Bp, int NTl,
                                            int nt0, int nstrips, int KC,
                                            _Float16* __restrict__ dst) {
    int total = KC * nstrips * 32;              // 16-half groups (32B each)
    for (int g = threadIdx.x; g < total; g += TPB) {
        int lane = g & 31;
        int rest = g >> 5;
        int sl = rest % nstrips;
        int kc = rest / nstrips;
        const _Float16* src = Bp + (((size_t)kc * NTl + nt0 + sl) * 32 + lane) * 16;
        _Float16* d = dst + (size_t)g * 16;
#if USE_ASYNC_LDS
        __builtin_amdgcn_global_load_async_to_lds_b128(
            (__attribute__((address_space(1))) v4i*)(src),
            (__attribute__((address_space(3))) v4i*)(d), 0, 0);
        __builtin_amdgcn_global_load_async_to_lds_b128(
            (__attribute__((address_space(1))) v4i*)(src + 8),
            (__attribute__((address_space(3))) v4i*)(d + 8), 0, 0);
#else
        *(v16h*)(d) = *(const v16h*)(src);
#endif
    }
}

// ---------------------------------------------------------------------------
__global__ void __launch_bounds__(TPB)
lstm_persistent(const float* __restrict__ x,
                const _Float16* __restrict__ Bp1, const _Float16* __restrict__ Bp2,
                const _Float16* __restrict__ Bp3, const _Float16* __restrict__ Wh,
                const float* __restrict__ bias1, const float* __restrict__ bias2,
                const float* __restrict__ bias3, const float* __restrict__ biasH,
                _Float16* __restrict__ act1, _Float16* __restrict__ act2,
                _Float16* __restrict__ act3, _Float16* __restrict__ h3pad,
                float* __restrict__ c1, float* __restrict__ c2, float* __restrict__ c3,
                float* __restrict__ h1f, float* __restrict__ h2f, float* __restrict__ h3f,
                float* __restrict__ gates, float* __restrict__ zbuf,
                float* __restrict__ out, int* __restrict__ bar) {
    extern __shared__ __align__(32) _Float16 smem[];

    int gtid = blockIdx.x * TPB + threadIdx.x;
    int wave = gtid >> 5;
    int lane = threadIdx.x & 31;
    int sl   = (threadIdx.x >> 6) & 1;          // local strip within WG

    // ---- one-time weight staging: global (L2) -> LDS ----
    int nt0 = 2 * blockIdx.x;
    stage_slice(Bp1, NT, nt0, 2, KC1, smem + LDS_L1);
    stage_slice(Bp2, NT, nt0, 2, KC2, smem + LDS_L2);
    stage_slice(Bp3, NT, nt0, 2, KC2, smem + LDS_L3);
    if (blockIdx.x < 8)                          // head strip nt == blockIdx.x
        stage_slice(Wh, 8, blockIdx.x, 1, KCH, smem + LDS_H);
#if USE_ASYNC_LDS
    wait_async_zero();
#endif
    __syncthreads();

    for (int t = 0; t < Ssz; ++t) {
        // ---- layer 1 ----
        gemm_layer(act1, KP1, smem + LDS_L1, sl, bias1, gates, KC1, wave, lane);
        gbar(bar);
        lstm_ew(gates, c1, h1f, act1, KP1, 3, act2, KP2, 3, gtid);
        gbar(bar);
        // ---- layer 2 ----
        gemm_layer(act2, KP2, smem + LDS_L2, sl, bias2, gates, KC2, wave, lane);
        gbar(bar);
        lstm_ew(gates, c2, h2f, act2, KP2, 403, act3, KP2, 3, gtid);
        gbar(bar);
        // ---- layer 3 ----
        gemm_layer(act3, KP2, smem + LDS_L3, sl, bias3, gates, KC2, wave, lane);
        gbar(bar);
        lstm_ew(gates, c3, h3f, act3, KP2, 403, h3pad, KPH, 0, gtid);
        if (t + 1 < Ssz) {                       // stage x_{t+1}
            for (int e = gtid; e < 192; e += NTHREADS) {
                int b = e / 3, d = e - b * 3;
                _Float16 xv = (_Float16)x[((t + 1) * 64 + b) * 3 + d];
                act1[b * KP1 + d] = xv;
                act2[b * KP2 + d] = xv;
                act3[b * KP2 + d] = xv;
            }
        }
        gbar(bar);
        // ---- MDN head GEMM: z(64x128) = h3pad(64x416) @ Wh, waves 0..31 ----
        if (wave < 32) {
            int nt = wave >> 2;
            int mt = wave & 3;
            v8f acc = {};
            v16h af = load_a_frag(h3pad, KPH, mt, 0, lane);
            for (int kc = 0; kc < KCH; ++kc) {
                v16h bf = *(const v16h*)(smem + LDS_H + (kc * 32 + lane) * 16);
                v16h an = af;
                if (kc + 1 < KCH) an = load_a_frag(h3pad, KPH, mt, kc + 1, lane);
                acc = __builtin_amdgcn_wmma_f32_16x16x32_f16(false, af, false, bf, (short)0, acc, false, false);
                af = an;
            }
            int n     = nt * 16 + (lane & 15);
            float bv  = biasH[n];
            int mbase = mt * 16 + ((lane >> 4) << 3);
#pragma unroll
            for (int r = 0; r < 8; ++r)
                zbuf[(mbase + r) * NHEAD + n] = acc[r] + bv;
        }
        gbar(bar);
        // ---- head nonlinearities: one thread per (b, t) row ----
        for (int b = gtid; b < 64; b += NTHREADS) {
            const float* z = zbuf + b * NHEAD;
            int row = b * Ssz + t;
            out[OFF_E + row] = 1.0f / (1.0f + __expf(z[0]));
            float mx = z[1];
#pragma unroll
            for (int j = 1; j < 20; ++j) mx = fmaxf(mx, z[1 + j]);
            float p[20], sum = 0.0f;
#pragma unroll
            for (int j = 0; j < 20; ++j) { p[j] = __expf(z[1 + j] - mx); sum += p[j]; }
            float inv = 1.0f / sum;
            float* pi = out + OFF_PI + row * 20;
            float* m1 = out + OFF_MU1 + row * 20;
            float* m2 = out + OFF_MU2 + row * 20;
            float* s1 = out + OFF_SIG1 + row * 20;
            float* s2 = out + OFF_SIG2 + row * 20;
            float* ro = out + OFF_RO + row * 20;
#pragma unroll
            for (int j = 0; j < 20; ++j) {
                pi[j] = p[j] * inv;
                m1[j] = z[21 + j];
                m2[j] = z[41 + j];
                s1[j] = __expf(z[61 + j]);
                s2[j] = __expf(z[81 + j]);
                ro[j] = tanhf(z[101 + j]);
            }
        }
        // no trailing barrier: zbuf is next written only after >=2 collective barriers
    }
    // final states
    for (int e = gtid; e < 25600; e += NTHREADS) {
        out[OFF_H1 + e] = h1f[e]; out[OFF_C1 + e] = c1[e];
        out[OFF_H2 + e] = h2f[e]; out[OFF_C2 + e] = c2[e];
        out[OFF_H3 + e] = h3f[e]; out[OFF_C3 + e] = c3[e];
    }
}

// ---------------------------------------------------------------------------
// Pack one layer's [Wih;Whh] (f32, row = gate) into f16 WMMA-B fragment order.
__global__ void prep_layer(const float* __restrict__ Wih, int KinW,
                           const float* __restrict__ Whh,
                           _Float16* __restrict__ Bp, int KC) {
    int total = KC * 51200;
    for (int e = blockIdx.x * blockDim.x + threadIdx.x; e < total;
         e += gridDim.x * blockDim.x) {
        int h    = e & 15;
        int lane = (e >> 4) & 31;
        int nt   = (e >> 9) % 100;
        int kc   = e / 51200;
        int n = nt * 16 + (lane & 15);
        int k = kc * 32 + ((lane & 16) ? 16 : 0) + h;
        float v = 0.0f;
        if (k < KinW)            v = Wih[n * KinW + k];
        else if (k < KinW + 400) v = Whh[n * 400 + (k - KinW)];
        Bp[e] = (_Float16)v;
    }
}

__device__ __forceinline__ void head_row(int r, const float* We, const float* Wpi,
                                         const float* Wm1, const float* Wm2,
                                         const float* Ws1, const float* Ws2,
                                         const float* Wro, const float** W, int* rr) {
    if (r == 0)       { *W = We;  *rr = 0; }
    else if (r < 21)  { *W = Wpi; *rr = r - 1; }
    else if (r < 41)  { *W = Wm1; *rr = r - 21; }
    else if (r < 61)  { *W = Wm2; *rr = r - 41; }
    else if (r < 81)  { *W = Ws1; *rr = r - 61; }
    else if (r < 101) { *W = Ws2; *rr = r - 81; }
    else              { *W = Wro; *rr = r - 101; }
}

// Fold head weights (hidden=[H3,H3,H3] => sum the three 400-wide blocks) and pack.
__global__ void prep_head(const float* __restrict__ We,  const float* __restrict__ be,
                          const float* __restrict__ Wpi, const float* __restrict__ bpi,
                          const float* __restrict__ Wm1, const float* __restrict__ bm1,
                          const float* __restrict__ Wm2, const float* __restrict__ bm2,
                          const float* __restrict__ Ws1, const float* __restrict__ bs1,
                          const float* __restrict__ Ws2, const float* __restrict__ bs2,
                          const float* __restrict__ Wro, const float* __restrict__ bro,
                          _Float16* __restrict__ Wh, float* __restrict__ biasH) {
    const int NW = KCH * 8 * 512;
    for (int i = blockIdx.x * blockDim.x + threadIdx.x; i < NW + NHEAD;
         i += gridDim.x * blockDim.x) {
        if (i < NW) {
            int e    = i;
            int h    = e & 15;
            int lane = (e >> 4) & 31;
            int nt   = (e >> 9) & 7;
            int kc   = e >> 12;
            int r = nt * 16 + (lane & 15);
            int k = kc * 32 + ((lane & 16) ? 16 : 0) + h;
            float v = 0.0f;
            if (r < 121 && k < 400) {
                const float* W; int rr;
                head_row(r, We, Wpi, Wm1, Wm2, Ws1, Ws2, Wro, &W, &rr);
                v = W[rr * 1200 + k] + W[rr * 1200 + 400 + k] + W[rr * 1200 + 800 + k];
            }
            Wh[e] = (_Float16)v;
        } else {
            int r = i - NW;
            float v = 0.0f;
            if (r == 0)       v = be[0];
            else if (r < 21)  v = bpi[r - 1];
            else if (r < 41)  v = bm1[r - 21];
            else if (r < 61)  v = bm2[r - 41];
            else if (r < 81)  v = bs1[r - 61];
            else if (r < 101) v = bs2[r - 81];
            else if (r < 121) v = bro[r - 101];
            biasH[r] = v;
        }
    }
}

// Initialize biases, activation staging (x_0 + initial states + zero pads),
// f32 c/h state buffers, and the global barrier.
__global__ void prep_misc(const float* __restrict__ x,
                          const float* __restrict__ h1in, const float* __restrict__ c1in,
                          const float* __restrict__ h2in, const float* __restrict__ c2in,
                          const float* __restrict__ h3in, const float* __restrict__ c3in,
                          const float* __restrict__ bih1, const float* __restrict__ bhh1,
                          const float* __restrict__ bih2, const float* __restrict__ bhh2,
                          const float* __restrict__ bih3, const float* __restrict__ bhh3,
                          float* __restrict__ bias1, float* __restrict__ bias2,
                          float* __restrict__ bias3,
                          _Float16* __restrict__ act1, _Float16* __restrict__ act2,
                          _Float16* __restrict__ act3, _Float16* __restrict__ h3pad,
                          float* __restrict__ c1, float* __restrict__ c2, float* __restrict__ c3,
                          float* __restrict__ h1f, float* __restrict__ h2f, float* __restrict__ h3f,
                          int* __restrict__ bar) {
    const int N_BIAS = 3 * G4H;
    const int N_A1 = 64 * KP1, N_A2 = 64 * KP2, N_A3 = 64 * KP2, N_HP = 64 * KPH;
    const int N_ST = 6 * 25600;
    const int TOT = N_BIAS + N_A1 + N_A2 + N_A3 + N_HP + N_ST + 2;
    for (int i = blockIdx.x * blockDim.x + threadIdx.x; i < TOT;
         i += gridDim.x * blockDim.x) {
        int e = i;
        if (e < N_BIAS) {
            int l = e / G4H, n = e - l * G4H;
            if (l == 0)      bias1[n] = bih1[n] + bhh1[n];
            else if (l == 1) bias2[n] = bih2[n] + bhh2[n];
            else             bias3[n] = bih3[n] + bhh3[n];
            continue;
        }
        e -= N_BIAS;
        if (e < N_A1) {
            int b = e / KP1, c = e - b * KP1;
            float v = 0.0f;
            if (c < 3)        v = x[b * 3 + c];
            else if (c < 403) v = h1in[b * 400 + c - 3];
            act1[e] = (_Float16)v; continue;
        }
        e -= N_A1;
        if (e < N_A2) {
            int b = e / KP2, c = e - b * KP2;
            float v = 0.0f;
            if (c < 3)                     v = x[b * 3 + c];
            else if (c >= 403 && c < 803)  v = h2in[b * 400 + c - 403];
            act2[e] = (_Float16)v; continue;
        }
        e -= N_A2;
        if (e < N_A3) {
            int b = e / KP2, c = e - b * KP2;
            float v = 0.0f;
            if (c < 3)                     v = x[b * 3 + c];
            else if (c >= 403 && c < 803)  v = h3in[b * 400 + c - 403];
            act3[e] = (_Float16)v; continue;
        }
        e -= N_A3;
        if (e < N_HP) {
            int b = e / KPH, c = e - b * KPH;
            h3pad[e] = (_Float16)((c < 400) ? h3in[b * 400 + c] : 0.0f);
            continue;
        }
        e -= N_HP;
        if (e < N_ST) {
            int which = e / 25600, j = e - which * 25600;
            switch (which) {
                case 0: c1[j] = c1in[j]; break;
                case 1: c2[j] = c2in[j]; break;
                case 2: c3[j] = c3in[j]; break;
                case 3: h1f[j] = h1in[j]; break;
                case 4: h2f[j] = h2in[j]; break;
                default: h3f[j] = h3in[j]; break;
            }
            continue;
        }
        e -= N_ST;
        bar[e] = 0;
    }
}

// ---------------------------------------------------------------------------
extern "C" void kernel_launch(void* const* d_in, const int* in_sizes, int n_in,
                              void* d_out, int out_size, void* d_ws, size_t ws_size,
                              hipStream_t stream) {
    const float* x    = (const float*)d_in[0];
    const float* h1in = (const float*)d_in[1];
    const float* c1in = (const float*)d_in[2];
    const float* h2in = (const float*)d_in[3];
    const float* c2in = (const float*)d_in[4];
    const float* h3in = (const float*)d_in[5];
    const float* c3in = (const float*)d_in[6];
    const float* Wih1 = (const float*)d_in[7];
    const float* Whh1 = (const float*)d_in[8];
    const float* bih1 = (const float*)d_in[9];
    const float* bhh1 = (const float*)d_in[10];
    const float* Wih2 = (const float*)d_in[11];
    const float* Whh2 = (const float*)d_in[12];
    const float* bih2 = (const float*)d_in[13];
    const float* bhh2 = (const float*)d_in[14];
    const float* Wih3 = (const float*)d_in[15];
    const float* Whh3 = (const float*)d_in[16];
    const float* bih3 = (const float*)d_in[17];
    const float* bhh3 = (const float*)d_in[18];
    const float* We  = (const float*)d_in[19]; const float* be  = (const float*)d_in[20];
    const float* Wpi = (const float*)d_in[21]; const float* bpi = (const float*)d_in[22];
    const float* Wm1 = (const float*)d_in[23]; const float* bm1 = (const float*)d_in[24];
    const float* Wm2 = (const float*)d_in[25]; const float* bm2 = (const float*)d_in[26];
    const float* Ws1 = (const float*)d_in[27]; const float* bs1 = (const float*)d_in[28];
    const float* Ws2 = (const float*)d_in[29]; const float* bs2 = (const float*)d_in[30];
    const float* Wro = (const float*)d_in[31]; const float* bro = (const float*)d_in[32];
    float* out = (float*)d_out;

    // ---- carve workspace (~8.2 MB total) ----
    char* w = (char*)d_ws;
    size_t off = 0;
    auto carve = [&](size_t bytes) -> void* {
        void* p = w + off;
        off += (bytes + 255) & ~(size_t)255;
        return p;
    };
    int*       bar   = (int*)carve(256);
    float*     bias1 = (float*)carve(G4H * 4);
    float*     bias2 = (float*)carve(G4H * 4);
    float*     bias3 = (float*)carve(G4H * 4);
    float*     biasH = (float*)carve(NHEAD * 4);
    float*     c1    = (float*)carve(25600 * 4);
    float*     c2    = (float*)carve(25600 * 4);
    float*     c3    = (float*)carve(25600 * 4);
    float*     h1f   = (float*)carve(25600 * 4);
    float*     h2f   = (float*)carve(25600 * 4);
    float*     h3f   = (float*)carve(25600 * 4);
    float*     gates = (float*)carve(64 * G4H * 4);
    float*     zbuf  = (float*)carve(64 * NHEAD * 4);
    _Float16*  act1  = (_Float16*)carve(64 * KP1 * 2);
    _Float16*  act2  = (_Float16*)carve(64 * KP2 * 2);
    _Float16*  act3  = (_Float16*)carve(64 * KP2 * 2);
    _Float16*  h3pad = (_Float16*)carve(64 * KPH * 2);
    _Float16*  Bp1   = (_Float16*)carve((size_t)KC1 * 51200 * 2);
    _Float16*  Bp2   = (_Float16*)carve((size_t)KC2 * 51200 * 2);
    _Float16*  Bp3   = (_Float16*)carve((size_t)KC2 * 51200 * 2);
    _Float16*  Wh    = (_Float16*)carve((size_t)KCH * 8 * 512 * 2);

    // ---- prep ----
    prep_layer<<<256, 256, 0, stream>>>(Wih1, 3,   Whh1, Bp1, KC1);
    prep_layer<<<256, 256, 0, stream>>>(Wih2, 403, Whh2, Bp2, KC2);
    prep_layer<<<256, 256, 0, stream>>>(Wih3, 403, Whh3, Bp3, KC2);
    prep_head<<<64, 256, 0, stream>>>(We, be, Wpi, bpi, Wm1, bm1, Wm2, bm2,
                                      Ws1, bs1, Ws2, bs2, Wro, bro, Wh, biasH);
    prep_misc<<<256, 256, 0, stream>>>(x, h1in, c1in, h2in, c2in, h3in, c3in,
                                       bih1, bhh1, bih2, bhh2, bih3, bhh3,
                                       bias1, bias2, bias3,
                                       act1, act2, act3, h3pad,
                                       c1, c2, c3, h1f, h2f, h3f, bar);

    // ---- persistent recurrent + fused-head kernel (146 KB dynamic LDS) ----
    (void)hipFuncSetAttribute((const void*)lstm_persistent,
                              hipFuncAttributeMaxDynamicSharedMemorySize, LDS_BYTES);
    lstm_persistent<<<NWG, TPB, LDS_BYTES, stream>>>(x, Bp1, Bp2, Bp3, Wh,
                                                     bias1, bias2, bias3, biasH,
                                                     act1, act2, act3, h3pad,
                                                     c1, c2, c3, h1f, h2f, h3f,
                                                     gates, zbuf, out, bar);
}
